// QUANT_FixedPoint_NoBN_BED_CLASSIFIER_1477468750319
// MI455X (gfx1250) — 8-bit matrix operand order repaired and checked by reading, compiled, not run on hardware
//
#include <hip/hip_runtime.h>
#include <stdint.h>


typedef int v8i __attribute__((ext_vector_type(8)));

#define EPSQ 1e-8f

__device__ __forceinline__ float pow2_scale(float maxv, float inv2bits) {
  return exp2f(ceilf(log2f(fmaxf(maxv, EPSQ)))) * inv2bits;
}

__global__ void k_init(float* g_max, float* g_scales,
                       unsigned int* qw_words, long qw_nwords) {
  long t = (long)blockIdx.x * blockDim.x + threadIdx.x;
  if (t < 16) g_max[t] = 0.0f;
  if (t == 0) g_scales[0] = 1.0f / 256.0f;
  long stride = (long)gridDim.x * blockDim.x;
  for (long i = t; i < qw_nwords; i += stride) qw_words[i] = 0u;
}

__global__ void k_weight_quant(const float* __restrict__ w,
                               signed char* __restrict__ qw,
                               float* __restrict__ wscale,
                               int K, int K_pad) {
  int oc = blockIdx.x;
  int tid = threadIdx.x;
  __shared__ float red[64];
  float m = 0.0f;
  for (int k = tid; k < K; k += 64) m = fmaxf(m, fabsf(w[(long)oc * K + k]));
  red[tid] = m;
  __syncthreads();
  if (tid == 0) {
    float mm = 0.0f;
    for (int i = 0; i < 64; ++i) mm = fmaxf(mm, red[i]);
    float s = pow2_scale(mm, 1.0f / 128.0f);
    wscale[oc] = s;
    red[0] = s;
  }
  __syncthreads();
  float s = red[0];
  for (int k = tid; k < K_pad; k += 64) {
    signed char q = 0;
    if (k < K) {
      float t = rintf(w[(long)oc * K + k] / s);
      t = fminf(fmaxf(t, -128.0f), 127.0f);
      q = (signed char)(int)t;
    }
    qw[(long)oc * K_pad + k] = q;
  }
}

__global__ void k_quant_u8x4(const float4* __restrict__ x,
                             uchar4* __restrict__ q,
                             const float* __restrict__ s_ptr, long nvec) {
  long t = (long)blockIdx.x * blockDim.x + threadIdx.x;
  if (t >= nvec) return;
  float inv = 1.0f / *s_ptr;
  float4 v = x[t];
  float a = fminf(fmaxf(rintf(v.x * inv), 0.0f), 255.0f);
  float b = fminf(fmaxf(rintf(v.y * inv), 0.0f), 255.0f);
  float c = fminf(fmaxf(rintf(v.z * inv), 0.0f), 255.0f);
  float d = fminf(fmaxf(rintf(v.w * inv), 0.0f), 255.0f);
  q[t] = make_uchar4((unsigned char)(int)a, (unsigned char)(int)b,
                     (unsigned char)(int)c, (unsigned char)(int)d);
}

__global__ void k_quant_pool_u8(const float* __restrict__ x,
                                unsigned char* __restrict__ q,
                                const float* __restrict__ s_ptr,
                                int BC, int H, int W) {
  int Ho = H >> 1, Wo = W >> 1;
  long nel = (long)BC * Ho * Wo;
  long t = (long)blockIdx.x * blockDim.x + threadIdx.x;
  if (t >= nel) return;
  int wo = (int)(t % Wo);
  long r = t / Wo;
  int ho = (int)(r % Ho);
  int bc = (int)(r / Ho);
  const float* p = x + ((long)bc * H + ho * 2) * W + wo * 2;
  float m = fmaxf(fmaxf(p[0], p[1]), fmaxf(p[W], p[W + 1]));
  float v = fminf(fmaxf(rintf(m / *s_ptr), 0.0f), 255.0f);
  q[t] = (unsigned char)(int)v;
}

__global__ void k_fin_scale(const float* __restrict__ maxv,
                            float* __restrict__ sout) {
  if (threadIdx.x == 0 && blockIdx.x == 0)
    *sout = pow2_scale(*maxv, 1.0f / 256.0f);
}

template <int KH, int MT>
__global__ void __launch_bounds__(32)
k_conv_wmma(const unsigned char* __restrict__ act,
            const signed char*  __restrict__ qW,
            const float* __restrict__ wscale,
            const float* __restrict__ bias,
            const float* __restrict__ s_in_ptr,
            float* __restrict__ out,
            unsigned int* __restrict__ maxv,
            int C, int H, int W, int O, int pad, int K, int K_pad) {
  constexpr int khw = KH * KH;
  const int lane = threadIdx.x;
  const int hi   = lane >> 4;
  const int n    = lane & 15;
  const int HW   = H * W;
  const int b    = blockIdx.y;
  const int rem  = blockIdx.x * 16 + n;
  const int oh   = rem / W;
  const int ow   = rem - oh * W;
  const int KT   = K_pad >> 6;
  const float s_in = *s_in_ptr;
  const unsigned char* actb = act + (long)b * C * HW;

  v8i acc[MT] = {};

  for (int kt = 0; kt < KT; ++kt) {
    v8i Bv;
#pragma unroll
    for (int v = 0; v < 8; ++v) {
      int k0 = kt * 64 + hi * 16 + (v & 3) * 4 + (v >> 2) * 32;
      unsigned int w32 = 0;
#pragma unroll
      for (int byi = 0; byi < 4; ++byi) {
        int k = k0 + byi;
        unsigned int val = 0;
        if (k < K) {
          int c  = k / khw;
          int r  = k - c * khw;
          int kh = r / KH;
          int kw = r - kh * KH;
          int ih = oh + kh - pad;
          int iw = ow + kw - pad;
          if (ih >= 0 && ih < H && iw >= 0 && iw < W)
            val = actb[(long)c * HW + ih * W + iw];
        }
        w32 |= val << (8 * byi);
      }
      Bv[v] = (int)w32;
    }
#pragma unroll
    for (int mt = 0; mt < MT; ++mt) {
      v8i Av;
      const signed char* wr =
          qW + (long)(mt * 16 + n) * K_pad + kt * 64 + hi * 16;
#pragma unroll
      for (int v = 0; v < 8; ++v) {
        int koff = (v & 3) * 4 + (v >> 2) * 32;
        Av[v] = *(const int*)(wr + koff);
      }
      acc[mt] = __builtin_amdgcn_wmma_i32_16x16x64_iu8(
          true, Av, false, Bv, acc[mt], false, false);
    }
  }

  float lmax = 0.0f;
#pragma unroll
  for (int mt = 0; mt < MT; ++mt) {
#pragma unroll
    for (int j = 0; j < 8; ++j) {
      int oc = mt * 16 + hi * 8 + j;
      if (oc < O) {
        float sw = wscale[oc] * s_in;
        float bq = rintf(bias[oc] / sw);
        bq = fminf(fmaxf(bq, -32768.0f), 32767.0f);
        float y = ((float)acc[mt][j] + bq) * sw;
        y = fmaxf(y, 0.0f);
        out[((long)b * O + oc) * HW + rem] = y;
        lmax = fmaxf(lmax, y);
      }
    }
  }
  __shared__ float red[32];
  red[lane] = lmax;
  __syncthreads();
  if (lane == 0) {
    float m = 0.0f;
    for (int i = 0; i < 32; ++i) m = fmaxf(m, red[i]);
    atomicMax(maxv, __float_as_uint(m));
  }
}

__global__ void k_avgpool(const unsigned char* __restrict__ in,
                          float* __restrict__ out,
                          const float* __restrict__ s_ptr,
                          unsigned int* __restrict__ maxv,
                          int BC, int HW) {
  int t = blockIdx.x * blockDim.x + threadIdx.x;
  if (t >= BC) return;
  const unsigned char* p = in + (long)t * HW;
  int sum = 0;
  for (int i = 0; i < HW; ++i) sum += p[i];
  float y = (*s_ptr) * (float)sum / (float)HW;
  out[t] = y;
  atomicMax(maxv, __float_as_uint(y));
}

__global__ void k_linear(const unsigned char* __restrict__ qx,
                         const signed char*  __restrict__ qw,
                         const float* __restrict__ wscale,
                         const float* __restrict__ bias,
                         const float* __restrict__ s_in_ptr,
                         float* __restrict__ out,
                         unsigned int* __restrict__ maxv,
                         int Bn, int O, int Kf, int K_pad, int doRelu) {
  int t = blockIdx.x * blockDim.x + threadIdx.x;
  if (t >= Bn * O) return;
  int b = t / O, of = t - b * O;
  int acc = 0;
  for (int k = 0; k < Kf; ++k)
    acc += (int)qx[b * Kf + k] * (int)qw[of * K_pad + k];
  float sw = (*s_in_ptr) * wscale[of];
  float bq = rintf(bias[of] / sw);
  bq = fminf(fmaxf(bq, -32768.0f), 32767.0f);
  float y = ((float)acc + bq) * sw;
  if (doRelu) {
    y = fmaxf(y, 0.0f);
    atomicMax(maxv, __float_as_uint(y));
  }
  out[t] = y;
}

static inline int opad(int o) { return (o + 15) & ~15; }
static inline int kpad(int k) { return (k + 63) & ~63; }

static void run_conv(hipStream_t stream, const unsigned char* act,
                     const signed char* qW, const float* wscale,
                     const float* bias, const float* s_in, float* out,
                     unsigned int* maxv,
                     int C, int O, int H, int W, int KH, int pad, int K) {
  dim3 grid((unsigned)(H * W / 16), 32);
  int K_pad = kpad(K);
  int MT = opad(O) >> 4;
#define CONV_CASE(KHv, MTv)                                                    \
  k_conv_wmma<KHv, MTv><<<grid, 32, 0, stream>>>(                              \
      act, qW, wscale, bias, s_in, out, maxv, C, H, W, O, pad, K, K_pad)
  if (KH == 3) {
    if      (MT == 1) CONV_CASE(3, 1);
    else if (MT == 2) CONV_CASE(3, 2);
    else              CONV_CASE(3, 4);
  } else {
    if      (MT == 1) CONV_CASE(1, 1);
    else if (MT == 2) CONV_CASE(1, 2);
    else              CONV_CASE(1, 4);
  }
#undef CONV_CASE
}

static void run_quant(hipStream_t stream, const float* x, unsigned char* q,
                      const float* s, long n) {
  long nvec = n >> 2;
  k_quant_u8x4<<<(unsigned)((nvec + 255) / 256), 256, 0, stream>>>(
      (const float4*)x, (uchar4*)q, s, nvec);
}

static void run_quant_pool(hipStream_t stream, const float* x,
                           unsigned char* q, const float* s,
                           int BC, int H, int W) {
  long nel = (long)BC * (H >> 1) * (W >> 1);
  k_quant_pool_u8<<<(unsigned)((nel + 255) / 256), 256, 0, stream>>>(
      x, q, s, BC, H, W);
}

extern "C" void kernel_launch(void* const* d_in, const int* in_sizes, int n_in,
                              void* d_out, int out_size, void* d_ws,
                              size_t ws_size, hipStream_t stream) {
  (void)in_sizes; (void)n_in; (void)out_size; (void)ws_size;

  const float* x = (const float*)d_in[0];
  const float* W14[14];
  const float* B14[14];
  for (int i = 0; i < 14; ++i) {
    W14[i] = (const float*)d_in[1 + 2 * i];
    B14[i] = (const float*)d_in[2 + 2 * i];
  }
  static const int LO[14] = {32,16,16,32,32,64,32,64,32,64,32,64,16,2};
  static const int LK[14] = {27,288,16,144,32,288,64,288,64,288,64,288,64,16};

  char* ws = (char*)d_ws;
  float* scal     = (float*)(ws + 0);
  float* g_max    = scal;
  float* g_scales = scal + 16;
  float* wsc      = (float*)(ws + 1024);
  signed char* qw = (signed char*)(ws + 8192);

  long qoff[14];
  long qtot = 0;
  for (int i = 0; i < 14; ++i) {
    qoff[i] = qtot;
    qtot += (long)opad(LO[i]) * kpad(LK[i]);
  }

  const long OFF_A = 131072;
  const long SZ_U8 = 51380224;
  unsigned char* uA = (unsigned char*)(ws + OFF_A);
  unsigned char* uB = uA + SZ_U8;
  float* F = (float*)(ws + OFF_A + 2 * SZ_U8);

  unsigned int* mx = (unsigned int*)g_max;

  k_init<<<64, 256, 0, stream>>>(g_max, g_scales, (unsigned int*)qw, qtot / 4);

  for (int i = 0; i < 14; ++i)
    k_weight_quant<<<LO[i], 64, 0, stream>>>(W14[i], qw + qoff[i],
                                             wsc + 64 * i, LK[i], kpad(LK[i]));

  run_quant(stream, x, uA, g_scales + 0, 32L * 3 * 224 * 224);

  run_conv(stream, uA, qw + qoff[0], wsc + 0,   B14[0], g_scales + 0, F, mx + 1,
           3, 32, 224, 224, 3, 1, 27);
  k_fin_scale<<<1, 1, 0, stream>>>(g_max + 1, g_scales + 1);
  run_quant_pool(stream, F, uB, g_scales + 1, 32 * 32, 224, 224);

  run_conv(stream, uB, qw + qoff[1], wsc + 64,  B14[1], g_scales + 1, F, mx + 2,
           32, 16, 112, 112, 3, 1, 288);
  k_fin_scale<<<1, 1, 0, stream>>>(g_max + 2, g_scales + 2);
  run_quant_pool(stream, F, uA, g_scales + 2, 32 * 16, 112, 112);

  run_conv(stream, uA, qw + qoff[2], wsc + 128, B14[2], g_scales + 2, F, mx + 3,
           16, 16, 56, 56, 1, 0, 16);
  k_fin_scale<<<1, 1, 0, stream>>>(g_max + 3, g_scales + 3);
  run_quant(stream, F, uB, g_scales + 3, 32L * 16 * 56 * 56);

  run_conv(stream, uB, qw + qoff[3], wsc + 192, B14[3], g_scales + 3, F, mx + 4,
           16, 32, 56, 56, 3, 1, 144);
  k_fin_scale<<<1, 1, 0, stream>>>(g_max + 4, g_scales + 4);
  run_quant(stream, F, uA, g_scales + 4, 32L * 32 * 56 * 56);

  run_conv(stream, uA, qw + qoff[4], wsc + 256, B14[4], g_scales + 4, F, mx + 5,
           32, 32, 56, 56, 1, 0, 32);
  k_fin_scale<<<1, 1, 0, stream>>>(g_max + 5, g_scales + 5);
  run_quant(stream, F, uB, g_scales + 5, 32L * 32 * 56 * 56);

  run_conv(stream, uB, qw + qoff[5], wsc + 320, B14[5], g_scales + 5, F, mx + 6,
           32, 64, 56, 56, 3, 1, 288);
  k_fin_scale<<<1, 1, 0, stream>>>(g_max + 6, g_scales + 6);
  run_quant_pool(stream, F, uA, g_scales + 6, 32 * 64, 56, 56);

  run_conv(stream, uA, qw + qoff[6], wsc + 384, B14[6], g_scales + 6, F, mx + 7,
           64, 32, 28, 28, 1, 0, 64);
  k_fin_scale<<<1, 1, 0, stream>>>(g_max + 7, g_scales + 7);
  run_quant(stream, F, uB, g_scales + 7, 32L * 32 * 28 * 28);

  run_conv(stream, uB, qw + qoff[7], wsc + 448, B14[7], g_scales + 7, F, mx + 8,
           32, 64, 28, 28, 3, 1, 288);
  k_fin_scale<<<1, 1, 0, stream>>>(g_max + 8, g_scales + 8);
  run_quant(stream, F, uA, g_scales + 8, 32L * 64 * 28 * 28);

  run_conv(stream, uA, qw + qoff[8], wsc + 512, B14[8], g_scales + 8, F, mx + 9,
           64, 32, 28, 28, 1, 0, 64);
  k_fin_scale<<<1, 1, 0, stream>>>(g_max + 9, g_scales + 9);
  run_quant(stream, F, uB, g_scales + 9, 32L * 32 * 28 * 28);

  run_conv(stream, uB, qw + qoff[9], wsc + 576, B14[9], g_scales + 9, F, mx + 10,
           32, 64, 28, 28, 3, 1, 288);
  k_fin_scale<<<1, 1, 0, stream>>>(g_max + 10, g_scales + 10);
  run_quant(stream, F, uA, g_scales + 10, 32L * 64 * 28 * 28);

  run_conv(stream, uA, qw + qoff[10], wsc + 640, B14[10], g_scales + 10, F, mx + 11,
           64, 32, 28, 28, 1, 0, 64);
  k_fin_scale<<<1, 1, 0, stream>>>(g_max + 11, g_scales + 11);
  run_quant(stream, F, uB, g_scales + 11, 32L * 32 * 28 * 28);

  run_conv(stream, uB, qw + qoff[11], wsc + 704, B14[11], g_scales + 11, F, mx + 12,
           32, 64, 28, 28, 3, 1, 288);
  k_fin_scale<<<1, 1, 0, stream>>>(g_max + 12, g_scales + 12);
  run_quant(stream, F, uA, g_scales + 12, 32L * 64 * 28 * 28);

  k_avgpool<<<8, 256, 0, stream>>>(uA, F, g_scales + 12, mx + 13, 32 * 64, 784);
  k_fin_scale<<<1, 1, 0, stream>>>(g_max + 13, g_scales + 13);
  run_quant(stream, F, uB, g_scales + 13, 2048);

  k_linear<<<2, 256, 0, stream>>>(uB, qw + qoff[12], wsc + 768, B14[12],
                                  g_scales + 13, F, mx + 14,
                                  32, 16, 64, kpad(64), 1);
  k_fin_scale<<<1, 1, 0, stream>>>(g_max + 14, g_scales + 14);
  run_quant(stream, F, uA, g_scales + 14, 512);

  k_linear<<<1, 64, 0, stream>>>(uA, qw + qoff[13], wsc + 832, B14[13],
                                 g_scales + 14, (float*)d_out, mx + 15,
                                 32, 2, 16, kpad(16), 0);
}
